// SparseMoELayer_678604833214
// MI455X (gfx1250) — matrix operand order repaired and checked by reading, not compiled, not run on hardware
//
#include <hip/hip_runtime.h>
#include <math.h>
#include <stdint.h>

// ---------------- problem constants ----------------
constexpr int E   = 16;        // experts
constexpr int D   = 2048;      // hidden
constexpr int F   = 4096;      // d_ff
constexpr int T   = 8192;      // tokens (4 * 2048)
constexpr int CAP = T;         // per-expert row capacity

// GEMM tiles
constexpr int MT  = 64;        // rows per block tile
constexpr int NT  = 128;       // cols per block tile
constexpr int KT  = 64;        // K per LDS stage (2 wmma k-steps of 32)
constexpr int LDK = 72;        // padded LDS K stride (bf16 elems); 144 B = 16B-aligned, bank-skewed

typedef __attribute__((ext_vector_type(16))) __bf16 v16bf;
typedef __attribute__((ext_vector_type(8)))  __bf16 bf16x8;
typedef __attribute__((ext_vector_type(8)))  float  v8f;

// ---------------- workspace layout (bytes, all 256-aligned) ----------------
constexpr size_t O_COUNTS = 0;                                        // 16 * 4
constexpr size_t O_ROWOF  = 256;                                      // 16*8192*4
constexpr size_t O_WROW   = O_ROWOF + (size_t)E * CAP * 4;            // 16384*4
constexpr size_t O_XBF    = O_WROW + (size_t)2 * T * 4;               // T*D bf16   (33.5 MB)
constexpr size_t O_W1T    = O_XBF + (size_t)T * D * 2;                // E*F*D bf16 (268 MB), [e][n][k]
constexpr size_t O_W2T    = O_W1T + (size_t)E * D * F * 2;            // E*D*F bf16 (268 MB), [e][n][k]
constexpr size_t O_H      = O_W2T + (size_t)E * D * F * 2;            // 2T*F bf16  (134 MB)
constexpr size_t O_OUTR   = O_H + (size_t)2 * T * F * 2;              // 2T*D f32   (134 MB)

// ---------------- CDNA5 async DMA helpers ----------------
// GLOBAL_LOAD_ASYNC_TO_LDS_B128: per-lane 16B global -> LDS, tracked by ASYNCcnt.
// VDST = VGPR with LDS byte offset (low 32 bits of a generic LDS pointer), VADDR = 64b global addr.
__device__ __forceinline__ void async_copy_b128(const void* gsrc, const __bf16* lds_dst) {
  asm volatile("global_load_async_to_lds_b128 %0, %1, off"
               :: "v"((unsigned)(uintptr_t)lds_dst), "v"(gsrc)
               : "memory");
}
#define ASYNC_WAIT(n) asm volatile("s_wait_asynccnt %0" :: "i"(n) : "memory")

// =====================================================================
// Layout kernels: fp32 -> bf16 (x) and fp32 -> bf16 transposed (weights)
// =====================================================================
__global__ __launch_bounds__(256) void cvt_x_kernel(const float* __restrict__ x,
                                                    __bf16* __restrict__ xbf) {
  size_t i = ((size_t)blockIdx.x * 256 + threadIdx.x) * 8;
  float4 a = *(const float4*)(x + i);
  float4 b = *(const float4*)(x + i + 4);
  bf16x8 v;
  v[0] = (__bf16)a.x; v[1] = (__bf16)a.y; v[2] = (__bf16)a.z; v[3] = (__bf16)a.w;
  v[4] = (__bf16)b.x; v[5] = (__bf16)b.y; v[6] = (__bf16)b.z; v[7] = (__bf16)b.w;
  *(bf16x8*)(xbf + i) = v;
}

// src[e][K][N] (fp32 row-major) -> dst[e][N][K] (bf16), tiled 32x32 via LDS
__global__ __launch_bounds__(256) void transpose_bf16_kernel(
    const float* __restrict__ src, __bf16* __restrict__ dst, int K, int N) {
  __shared__ float tile[32][33];
  const int e = blockIdx.z;
  const float* s = src + (size_t)e * K * N;
  __bf16* d = dst + (size_t)e * K * N;
  const int k0 = blockIdx.y * 32, n0 = blockIdx.x * 32;
  const int tx = threadIdx.x & 31, ty = threadIdx.x >> 5;   // 32 x 8
#pragma unroll
  for (int j = 0; j < 32; j += 8)
    tile[ty + j][tx] = s[(size_t)(k0 + ty + j) * N + n0 + tx];
  __syncthreads();
#pragma unroll
  for (int j = 0; j < 32; j += 8)
    d[(size_t)(n0 + ty + j) * K + k0 + tx] = (__bf16)tile[tx][ty + j];
}

// =====================================================================
// Router: logits = x @ router_w, top-2, softmax, per-expert row lists
// one wave (32 lanes) per token
// =====================================================================
__global__ __launch_bounds__(256) void router_kernel(
    const float* __restrict__ x, const float* __restrict__ rw,
    int* __restrict__ counts, int* __restrict__ row_of,
    float* __restrict__ w_row) {
  const int wid  = threadIdx.x >> 5;
  const int lane = threadIdx.x & 31;
  const int t    = blockIdx.x * 8 + wid;
  const float* xr = x + (size_t)t * D;

  float acc[E];
#pragma unroll
  for (int e = 0; e < E; ++e) acc[e] = 0.0f;
  for (int j = lane; j < D; j += 32) {
    const float xv = xr[j];
    const float* wr = rw + (size_t)j * E;
#pragma unroll
    for (int e = 0; e < E; ++e) acc[e] += xv * wr[e];
  }
#pragma unroll
  for (int e = 0; e < E; ++e) {
#pragma unroll
    for (int m = 16; m >= 1; m >>= 1) acc[e] += __shfl_xor(acc[e], m, 32);
  }

  if (lane == 0) {
    float b0 = -3.4e38f, b1 = -3.4e38f;
    int i0 = 0, i1 = 0;
#pragma unroll
    for (int e = 0; e < E; ++e) {           // strict > : lower index wins ties (lax.top_k)
      float v = acc[e];
      if (v > b0)      { b1 = b0; i1 = i0; b0 = v; i0 = e; }
      else if (v > b1) { b1 = v;  i1 = e; }
    }
    const float e1  = __expf(b1 - b0);
    const float inv = 1.0f / (1.0f + e1);
    const int r0 = 2 * t, r1 = 2 * t + 1;
    w_row[r0] = inv;
    w_row[r1] = e1 * inv;
    int s0 = atomicAdd(&counts[i0], 1); row_of[i0 * CAP + s0] = r0;
    int s1 = atomicAdd(&counts[i1], 1); row_of[i1 * CAP + s1] = r1;
  }
}

// =====================================================================
// WMMA fragment loaders (ISA 7.12.2 layouts, wave32, bf16 16x16x32)
// =====================================================================
// A 16x32: lane holds row M=lane&15; elems 0..7 -> K=kb+0..7, elems 8..15 -> K=16+kb+0..7, kb=(lane>=16)*8
__device__ __forceinline__ v16bf frag_a(const __bf16* row_base, int lane, int kk) {
  const __bf16* p = row_base + (lane & 15) * LDK + kk + ((lane >> 4) << 3);
  bf16x8 lo = *(const bf16x8*)p;
  bf16x8 hi = *(const bf16x8*)(p + 16);
  v16bf f;
#pragma unroll
  for (int i = 0; i < 8; ++i) { f[i] = lo[i]; f[i + 8] = hi[i]; }
  return f;
}
// B 32x16 (LDS [n][k]): lane holds col N=lane&15; elems 0..7 -> K=kb+0..7, elems 8..15 -> K=16+kb+0..7, kb=(lane>=16)*8
__device__ __forceinline__ v16bf frag_b(const __bf16* col_base, int lane, int kk) {
  const __bf16* p = col_base + (lane & 15) * LDK + kk + ((lane >> 4) << 3);
  bf16x8 lo = *(const bf16x8*)p;
  bf16x8 hi = *(const bf16x8*)(p + 16);
  v16bf f;
#pragma unroll
  for (int i = 0; i < 8; ++i) { f[i] = lo[i]; f[i + 8] = hi[i]; }
  return f;
}

// =====================================================================
// FFN1: h[r,:] = silu(xbf[token(r)] @ W1T[e]^T)   async-DMA double-buffered
// grid = (F/NT, T/MT, E); block = 256 (8 waves, 4(M) x 2(N))
// =====================================================================
__global__ __launch_bounds__(256) void ffn1_kernel(
    const __bf16* __restrict__ xbf, const __bf16* __restrict__ W1T,
    const int* __restrict__ counts, const int* __restrict__ row_of,
    __bf16* __restrict__ h) {
  const int e   = blockIdx.z;
  const int cnt = counts[e];
  const int m_base = blockIdx.y * MT;
  if (m_base >= cnt) return;
  const int n_base = blockIdx.x * NT;

  __shared__ __align__(16) __bf16 sA[2][MT][LDK];
  __shared__ __align__(16) __bf16 sB[2][NT][LDK];
  __shared__ int s_row[MT];
  __shared__ int s_src[MT];

  const int tid = threadIdx.x, lane = tid & 31, wid = tid >> 5;
  if (tid < MT) {
    int m = m_base + tid;
    int r = row_of[e * CAP + ((m < cnt) ? m : 0)];
    s_row[tid] = (m < cnt) ? r : -1;
    s_src[tid] = r >> 1;               // token id
  }
  __syncthreads();

  // per-thread DMA slots: 16B chunks; r0 in [0,32), cc in {0,8,..,56} bf16 elems
  const int r0 = tid >> 3, cc = (tid & 7) << 3;
  const __bf16* gA0 = xbf + (size_t)s_src[r0]      * D + cc;   // A rows r0, r0+32
  const __bf16* gA1 = xbf + (size_t)s_src[r0 + 32] * D + cc;
  const __bf16* gB0 = W1T + (size_t)e * F * D + (size_t)(n_base + r0) * D + cc;  // B rows r0+32*i

  const int wm = wid & 3, wn = wid >> 2;
  v8f acc[4];
#pragma unroll
  for (int a = 0; a < 4; ++a)
#pragma unroll
    for (int i = 0; i < 8; ++i) acc[a][i] = 0.0f;

  auto issue_stage = [&](int k0, int buf) {
    async_copy_b128(gA0 + k0, &sA[buf][r0][cc]);
    async_copy_b128(gA1 + k0, &sA[buf][r0 + 32][cc]);
#pragma unroll
    for (int i = 0; i < 4; ++i)
      async_copy_b128(gB0 + (size_t)(32 * i) * D + k0, &sB[buf][r0 + 32 * i][cc]);
  };

  constexpr int NK = D / KT;           // 32 stages
  issue_stage(0, 0);
  for (int s = 0; s < NK; ++s) {
    const int buf = s & 1;
    if (s + 1 < NK) { issue_stage((s + 1) * KT, buf ^ 1); ASYNC_WAIT(6); }
    else            { ASYNC_WAIT(0); }
    __syncthreads();                   // stage s resident for all waves
#pragma unroll
    for (int kk = 0; kk < KT; kk += 32) {
      v16bf a = frag_a(&sA[buf][wm * 16][0], lane, kk);
#pragma unroll
      for (int t4 = 0; t4 < 4; ++t4) {
        v16bf b = frag_b(&sB[buf][wn * 64 + t4 * 16][0], lane, kk);
        acc[t4] = __builtin_amdgcn_wmma_f32_16x16x32_bf16(
            false, a, false, b, (short)0, acc[t4], false, false);
      }
    }
    __syncthreads();                   // buffer free for stage s+2 DMA
  }

  // epilogue: SiLU, store bf16 to h[r]
#pragma unroll
  for (int t4 = 0; t4 < 4; ++t4) {
    int n = n_base + wn * 64 + t4 * 16 + (lane & 15);
#pragma unroll
    for (int j = 0; j < 8; ++j) {
      int m = wm * 16 + j + ((lane >> 4) << 3);
      int r = s_row[m];
      if (r >= 0) {
        float v = acc[t4][j];
        h[(size_t)r * F + n] = (__bf16)(v / (1.0f + __expf(-v)));
      }
    }
  }
}

// =====================================================================
// FFN2: out_rows[r,:] = w_row[r] * (h[r,:] @ W2T[e]^T)
// grid = (D/NT, T/MT, E); block = 256
// =====================================================================
__global__ __launch_bounds__(256) void ffn2_kernel(
    const __bf16* __restrict__ h, const __bf16* __restrict__ W2T,
    const int* __restrict__ counts, const int* __restrict__ row_of,
    const float* __restrict__ w_row, float* __restrict__ out_rows) {
  const int e   = blockIdx.z;
  const int cnt = counts[e];
  const int m_base = blockIdx.y * MT;
  if (m_base >= cnt) return;
  const int n_base = blockIdx.x * NT;

  __shared__ __align__(16) __bf16 sA[2][MT][LDK];
  __shared__ __align__(16) __bf16 sB[2][NT][LDK];
  __shared__ int   s_row[MT];
  __shared__ int   s_src[MT];
  __shared__ float s_w[MT];

  const int tid = threadIdx.x, lane = tid & 31, wid = tid >> 5;
  if (tid < MT) {
    int m = m_base + tid;
    int r = row_of[e * CAP + ((m < cnt) ? m : 0)];
    s_row[tid] = (m < cnt) ? r : -1;
    s_src[tid] = r;
    s_w[tid]   = (m < cnt) ? w_row[r] : 0.0f;
  }
  __syncthreads();

  const int r0 = tid >> 3, cc = (tid & 7) << 3;
  const __bf16* gA0 = h + (size_t)s_src[r0]      * F + cc;
  const __bf16* gA1 = h + (size_t)s_src[r0 + 32] * F + cc;
  const __bf16* gB0 = W2T + (size_t)e * D * F + (size_t)(n_base + r0) * F + cc;

  const int wm = wid & 3, wn = wid >> 2;
  v8f acc[4];
#pragma unroll
  for (int a = 0; a < 4; ++a)
#pragma unroll
    for (int i = 0; i < 8; ++i) acc[a][i] = 0.0f;

  auto issue_stage = [&](int k0, int buf) {
    async_copy_b128(gA0 + k0, &sA[buf][r0][cc]);
    async_copy_b128(gA1 + k0, &sA[buf][r0 + 32][cc]);
#pragma unroll
    for (int i = 0; i < 4; ++i)
      async_copy_b128(gB0 + (size_t)(32 * i) * F + k0, &sB[buf][r0 + 32 * i][cc]);
  };

  constexpr int NK = F / KT;           // 64 stages
  issue_stage(0, 0);
  for (int s = 0; s < NK; ++s) {
    const int buf = s & 1;
    if (s + 1 < NK) { issue_stage((s + 1) * KT, buf ^ 1); ASYNC_WAIT(6); }
    else            { ASYNC_WAIT(0); }
    __syncthreads();
#pragma unroll
    for (int kk = 0; kk < KT; kk += 32) {
      v16bf a = frag_a(&sA[buf][wm * 16][0], lane, kk);
#pragma unroll
      for (int t4 = 0; t4 < 4; ++t4) {
        v16bf b = frag_b(&sB[buf][wn * 64 + t4 * 16][0], lane, kk);
        acc[t4] = __builtin_amdgcn_wmma_f32_16x16x32_bf16(
            false, a, false, b, (short)0, acc[t4], false, false);
      }
    }
    __syncthreads();
  }

  // epilogue: scale by routing weight, per-row fp32 store (no atomics -> deterministic)
#pragma unroll
  for (int t4 = 0; t4 < 4; ++t4) {
    int n = n_base + wn * 64 + t4 * 16 + (lane & 15);
#pragma unroll
    for (int j = 0; j < 8; ++j) {
      int m = wm * 16 + j + ((lane >> 4) << 3);
      int r = s_row[m];
      if (r >= 0) out_rows[(size_t)r * D + n] = acc[t4][j] * s_w[m];
    }
  }
}

// =====================================================================
// Combine: out[t,:] = out_rows[2t,:] + out_rows[2t+1,:]
// =====================================================================
__global__ __launch_bounds__(256) void combine_kernel(
    const float* __restrict__ out_rows, float* __restrict__ out) {
  size_t i = ((size_t)blockIdx.x * 256 + threadIdx.x) * 4;   // over T*D
  size_t t = i >> 11, d = i & 2047;
  float4 a = *(const float4*)(out_rows + (t * 2) * (size_t)D + d);
  float4 b = *(const float4*)(out_rows + (t * 2 + 1) * (size_t)D + d);
  float4 o; o.x = a.x + b.x; o.y = a.y + b.y; o.z = a.z + b.z; o.w = a.w + b.w;
  *(float4*)(out + i) = o;
}

// =====================================================================
extern "C" void kernel_launch(void* const* d_in, const int* in_sizes, int n_in,
                              void* d_out, int out_size, void* d_ws, size_t ws_size,
                              hipStream_t stream) {
  const float* x  = (const float*)d_in[0];   // [4,2048,2048]
  const float* rw = (const float*)d_in[1];   // [2048,16]
  const float* W1 = (const float*)d_in[2];   // [16,2048,4096]
  const float* W2 = (const float*)d_in[3];   // [16,4096,2048]
  float* out = (float*)d_out;

  char* ws = (char*)d_ws;
  int*    counts   = (int*)(ws + O_COUNTS);
  int*    row_of   = (int*)(ws + O_ROWOF);
  float*  w_row    = (float*)(ws + O_WROW);
  __bf16* xbf      = (__bf16*)(ws + O_XBF);
  __bf16* W1T      = (__bf16*)(ws + O_W1T);
  __bf16* W2T      = (__bf16*)(ws + O_W2T);
  __bf16* h        = (__bf16*)(ws + O_H);
  float*  out_rows = (float*)(ws + O_OUTR);

  hipMemsetAsync(counts, 0, E * sizeof(int), stream);
  cvt_x_kernel<<<(size_t)T * D / (8 * 256), 256, 0, stream>>>(x, xbf);
  transpose_bf16_kernel<<<dim3(F / 32, D / 32, E), 256, 0, stream>>>(W1, W1T, D, F);
  transpose_bf16_kernel<<<dim3(D / 32, F / 32, E), 256, 0, stream>>>(W2, W2T, F, D);
  router_kernel<<<T / 8, 256, 0, stream>>>(x, rw, counts, row_of, w_row);
  ffn1_kernel<<<dim3(F / NT, T / MT, E), 256, 0, stream>>>(xbf, W1T, counts, row_of, h);
  ffn2_kernel<<<dim3(D / NT, T / MT, E), 256, 0, stream>>>(h, W2T, counts, row_of, w_row, out_rows);
  combine_kernel<<<(size_t)T * D / (4 * 256), 256, 0, stream>>>(out_rows, out);
}
